// TopKSAE_3152505995467
// MI455X (gfx1250) — matrix operand order repaired and checked by reading, not compiled, not run on hardware
//
#include <hip/hip_runtime.h>
#include <hip/hip_bf16.h>
#include <stdint.h>

#define D_IN     768
#define D_SAE    24576
#define TOPK     64
#define N_TOK    8192

#define BM 128
#define BN 128
#define BK 32
#define LP 40   // LDS pitch in bf16 elems (32 + 8 pad -> conflict-free b128 reads)
#define KSTEPS (D_IN / BK)   // 24

typedef __bf16 v16bf __attribute__((ext_vector_type(16)));
typedef float  v8f   __attribute__((ext_vector_type(8)));

static __device__ __forceinline__ unsigned short f32_to_bf16(float f) {
    unsigned int u = __float_as_uint(f);
    u = u + 0x7fffu + ((u >> 16) & 1u);   // round-to-nearest-even
    return (unsigned short)(u >> 16);
}
static __device__ __forceinline__ unsigned int orderKey(float f) {
    unsigned int u = __float_as_uint(f);
    return (u & 0x80000000u) ? ~u : (u | 0x80000000u);
}

// CDNA5 async global->LDS copy (ASYNCcnt-tracked, no VGPR round-trip).
// vdst = LDS byte address, vaddr = 64-bit global address (GV mode).
static __device__ __forceinline__ void async_b128(unsigned lds, const void* g) {
    asm volatile("global_load_async_to_lds_b128 %0, %1, off"
                 :: "v"(lds), "v"((unsigned long long)(uintptr_t)g)
                 : "memory");
}
static __device__ __forceinline__ void wait_async_le4() {
    asm volatile("s_wait_asynccnt 0x4" ::: "memory");
}
static __device__ __forceinline__ void wait_async_0() {
    asm volatile("s_wait_asynccnt 0x0" ::: "memory");
}

// ---------------- prep: xb = bf16(x - b_dec), wb = bf16(W_enc) ----------------
__global__ __launch_bounds__(256) void k_prep_x(const float* __restrict__ x,
                                                const float* __restrict__ b_dec,
                                                unsigned short* __restrict__ xb) {
    size_t i = ((size_t)blockIdx.x * blockDim.x + threadIdx.x) * 4;
    if (i >= (size_t)N_TOK * D_IN) return;
    int c = (int)(i % D_IN);
    float4 v = *(const float4*)(x + i);
    float4 b = *(const float4*)(b_dec + c);
    unsigned int p0 = (unsigned int)f32_to_bf16(v.x - b.x) |
                      ((unsigned int)f32_to_bf16(v.y - b.y) << 16);
    unsigned int p1 = (unsigned int)f32_to_bf16(v.z - b.z) |
                      ((unsigned int)f32_to_bf16(v.w - b.w) << 16);
    uint2 o; o.x = p0; o.y = p1;
    *(uint2*)(xb + i) = o;
}

__global__ __launch_bounds__(256) void k_prep_w(const float* __restrict__ w,
                                                unsigned short* __restrict__ wb) {
    size_t i = ((size_t)blockIdx.x * blockDim.x + threadIdx.x) * 4;
    if (i >= (size_t)D_SAE * D_IN) return;
    float4 v = *(const float4*)(w + i);
    unsigned int p0 = (unsigned int)f32_to_bf16(v.x) | ((unsigned int)f32_to_bf16(v.y) << 16);
    unsigned int p1 = (unsigned int)f32_to_bf16(v.z) | ((unsigned int)f32_to_bf16(v.w) << 16);
    uint2 o; o.x = p0; o.y = p1;
    *(uint2*)(wb + i) = o;
}

// ---------------- W_dec [D_IN][D_SAE] -> Wt [D_SAE][D_IN] ----------------
__global__ __launch_bounds__(256) void k_transpose(const float* __restrict__ in,
                                                   float* __restrict__ out) {
    __shared__ float tile[32][33];
    int bs = blockIdx.x * 32;   // along D_SAE
    int bd = blockIdx.y * 32;   // along D_IN
    int tx = threadIdx.x, ty = threadIdx.y;
    for (int r = ty; r < 32; r += 8)
        tile[r][tx] = in[(size_t)(bd + r) * D_SAE + bs + tx];
    __syncthreads();
    for (int r = ty; r < 32; r += 8)
        out[(size_t)(bs + r) * D_IN + bd + tx] = tile[tx][r];
}

// ---------------- encode GEMM: pre = xb @ wb^T + b_enc (bf16 WMMA) ----------------
// Double-buffered LDS, staged with GLOBAL_LOAD_ASYNC_TO_LDS_B128 (ASYNCcnt).
__global__ __launch_bounds__(256) void k_encode_gemm(const unsigned short* __restrict__ xb,
                                                     const unsigned short* __restrict__ wb,
                                                     const float* __restrict__ b_enc,
                                                     float* __restrict__ pre) {
    __shared__ unsigned short As[2][BM * LP];
    __shared__ unsigned short Bs[2][BN * LP];

    const int tid  = threadIdx.x;
    const int lane = tid & 31;
    const int wv   = tid >> 5;          // wave 0..7
    const int wm   = (wv >> 2) * 64;    // wave tile M origin: 0 / 64
    const int wn   = (wv & 3)  * 32;    // wave tile N origin: 0/32/64/96
    const int kh   = lane >> 4;         // half-wave selector
    const int lr   = lane & 15;

    const int bm = blockIdx.y * BM;
    const int bn = blockIdx.x * BN;

    // --- per-thread staging addresses: 2 chunks of A + 2 chunks of B per stage ---
    const int r0 = tid >> 2;            // rows 0..63
    const int r1 = r0 + 64;             // rows 64..127
    const int cc = (tid & 3) * 8;       // k-chunk within BK (8 bf16 = 16B)

    const unsigned short* gA0 = xb + (size_t)(bm + r0) * D_IN + cc;
    const unsigned short* gA1 = xb + (size_t)(bm + r1) * D_IN + cc;
    const unsigned short* gB0 = wb + (size_t)(bn + r0) * D_IN + cc;
    const unsigned short* gB1 = wb + (size_t)(bn + r1) * D_IN + cc;

    unsigned lA0[2], lA1[2], lB0[2], lB1[2];
    #pragma unroll
    for (int b = 0; b < 2; ++b) {
        lA0[b] = (unsigned)(uintptr_t)&As[b][r0 * LP + cc];
        lA1[b] = (unsigned)(uintptr_t)&As[b][r1 * LP + cc];
        lB0[b] = (unsigned)(uintptr_t)&Bs[b][r0 * LP + cc];
        lB1[b] = (unsigned)(uintptr_t)&Bs[b][r1 * LP + cc];
    }

    auto issue = [&](int buf, int k0) {
        async_b128(lA0[buf], gA0 + k0);
        async_b128(lA1[buf], gA1 + k0);
        async_b128(lB0[buf], gB0 + k0);
        async_b128(lB1[buf], gB1 + k0);
    };

    v8f acc[4][2];
    v8f zero = {0.f, 0.f, 0.f, 0.f, 0.f, 0.f, 0.f, 0.f};
    for (int i = 0; i < 4; ++i)
        for (int j = 0; j < 2; ++j) acc[i][j] = zero;

    union FragU { uint4 q[2]; v16bf v; };

    issue(0, 0);                               // prologue: stage 0 -> buf 0
    for (int kk = 0; kk < KSTEPS; ++kk) {
        const int cur = kk & 1;
        if (kk + 1 < KSTEPS) {
            issue(cur ^ 1, (kk + 1) * BK);     // prefetch next stage
            wait_async_le4();                  // in-order completion: current stage landed
        } else {
            wait_async_0();
        }
        __syncthreads();                       // all waves' copies visible

        // A fragment: lanes 0-15 hold K[0..7],K[16..23]; lanes 16-31 K[8..15],K[24..31]
        v16bf a[4];
        #pragma unroll
        for (int i = 0; i < 4; ++i) {
            int row = wm + i * 16 + lr;
            FragU u;
            u.q[0] = *(const uint4*)&As[cur][row * LP + 8 * kh];
            u.q[1] = *(const uint4*)&As[cur][row * LP + 16 + 8 * kh];
            a[i] = u.v;
        }
        // B fragment: lanes 0-15 hold K[0..7],K[16..23]; lanes 16-31 K[8..15],K[24..31]
        v16bf b[2];
        #pragma unroll
        for (int j = 0; j < 2; ++j) {
            int row = wn + j * 16 + lr;
            FragU u;
            u.q[0] = *(const uint4*)&Bs[cur][row * LP + 8 * kh];
            u.q[1] = *(const uint4*)&Bs[cur][row * LP + 16 + 8 * kh];
            b[j] = u.v;
        }
        #pragma unroll
        for (int i = 0; i < 4; ++i)
            #pragma unroll
            for (int j = 0; j < 2; ++j)
                acc[i][j] = __builtin_amdgcn_wmma_f32_16x16x32_bf16(
                    false, a[i], false, b[j], (short)0, acc[i][j], false, false);
        __syncthreads();                       // reads done before buffer reuse
    }

    // epilogue: D element (vgpr v, lane l) -> M = v + 8*(l/16), N = l%16
    #pragma unroll
    for (int j = 0; j < 2; ++j) {
        int col = bn + wn + j * 16 + lr;
        float be = b_enc[col];
        #pragma unroll
        for (int i = 0; i < 4; ++i) {
            int rb = bm + wm + i * 16 + 8 * kh;
            #pragma unroll
            for (int v = 0; v < 8; ++v)
                pre[(size_t)(rb + v) * D_SAE + col] = acc[i][j][v] + be;
        }
    }
}

// ---------------- per-row radix-select top-64, rewrite acts in place ----------------
__global__ __launch_bounds__(256) void k_topk(float* __restrict__ acts,
                                              int* __restrict__ top_idx,
                                              float* __restrict__ top_val) {
    __shared__ unsigned int hist[256];
    __shared__ unsigned int s_prefix, s_krem, s_neq, s_nsel;
    __shared__ int   sIdx[TOPK];
    __shared__ float sVal[TOPK];

    const int tid = threadIdx.x;
    const int n   = blockIdx.x;
    float* __restrict__ row = acts + (size_t)n * D_SAE;

    if (tid == 0) { s_prefix = 0u; s_krem = TOPK; s_neq = 0u; s_nsel = 0u; }

    for (int pass = 0; pass < 4; ++pass) {
        const int sh = 24 - pass * 8;
        hist[tid] = 0u;
        __syncthreads();
        const unsigned int mask = pass ? (0xFFFFFFFFu << (sh + 8)) : 0u;
        const unsigned int pref = s_prefix;
        for (int s = tid; s < D_SAE; s += 256) {
            unsigned int k = orderKey(row[s]);
            if ((k & mask) == pref) atomicAdd(&hist[(k >> sh) & 255u], 1u);
        }
        __syncthreads();
        if (tid == 0) {
            unsigned int cum = 0, krem = s_krem, pbase = s_prefix;
            for (int b = 255; b >= 0; --b) {
                unsigned int h = hist[b];
                if (cum + h >= krem) {
                    s_krem   = krem - cum;
                    s_prefix = pbase | ((unsigned int)b << sh);
                    break;
                }
                cum += h;
            }
        }
        __syncthreads();
    }

    const unsigned int T = s_prefix;        // key of 64th-largest element
    const unsigned int needEq = s_krem;     // equals-at-threshold still needed
    for (int s = tid; s < D_SAE; s += 256) {
        float f = row[s];
        unsigned int k = orderKey(f);
        bool sel = false;
        if (k > T) sel = true;
        else if (k == T) { unsigned int e = atomicAdd(&s_neq, 1u); sel = (e < needEq); }
        float rv = f > 0.0f ? f : 0.0f;
        row[s] = sel ? rv : 0.0f;
        if (sel) {
            unsigned int slot = atomicAdd(&s_nsel, 1u);
            sIdx[slot] = s; sVal[slot] = rv;
        }
    }
    __syncthreads();

    // bitonic sort the 64-entry compact list by index (deterministic replay order)
    for (int kk = 2; kk <= TOPK; kk <<= 1) {
        for (int j = kk >> 1; j > 0; j >>= 1) {
            if (tid < TOPK) {
                int ixj = tid ^ j;
                if (ixj > tid) {
                    bool up = ((tid & kk) == 0);
                    int a = sIdx[tid], b = sIdx[ixj];
                    if ((a > b) == up) {
                        sIdx[tid] = b; sIdx[ixj] = a;
                        float va = sVal[tid], vb = sVal[ixj];
                        sVal[tid] = vb; sVal[ixj] = va;
                    }
                }
            }
            __syncthreads();
        }
    }
    if (tid < TOPK) {
        top_idx[(size_t)n * TOPK + tid] = sIdx[tid];
        top_val[(size_t)n * TOPK + tid] = sVal[tid];
    }
}

// ---------------- sparse decode: recon = b_dec + sum val_j * Wt[idx_j,:] ----------------
__global__ __launch_bounds__(256) void k_decode(const float* __restrict__ Wt,
                                                const float* __restrict__ b_dec,
                                                const int* __restrict__ top_idx,
                                                const float* __restrict__ top_val,
                                                float* __restrict__ recon) {
    __shared__ int   sIdx[TOPK];
    __shared__ float sVal[TOPK];
    const int n = blockIdx.x, tid = threadIdx.x;
    if (tid < TOPK) {
        sIdx[tid] = top_idx[(size_t)n * TOPK + tid];
        sVal[tid] = top_val[(size_t)n * TOPK + tid];
    }
    __syncthreads();
    float a0 = b_dec[tid], a1 = b_dec[tid + 256], a2 = b_dec[tid + 512];
    #pragma unroll 4
    for (int j = 0; j < TOPK; ++j) {
        const float* w = Wt + (size_t)sIdx[j] * D_IN;
        float v = sVal[j];
        a0 += v * w[tid]; a1 += v * w[tid + 256]; a2 += v * w[tid + 512];
    }
    float* r = recon + (size_t)n * D_IN;
    r[tid] = a0; r[tid + 256] = a1; r[tid + 512] = a2;
}

// ---------------- launch ----------------
extern "C" void kernel_launch(void* const* d_in, const int* in_sizes, int n_in,
                              void* d_out, int out_size, void* d_ws, size_t ws_size,
                              hipStream_t stream) {
    (void)in_sizes; (void)n_in; (void)out_size; (void)ws_size;
    const float* x     = (const float*)d_in[0];
    const float* W_enc = (const float*)d_in[1];
    const float* b_enc = (const float*)d_in[2];
    const float* W_dec = (const float*)d_in[3];
    const float* b_dec = (const float*)d_in[4];

    float* recon = (float*)d_out;
    float* acts  = recon + (size_t)N_TOK * D_IN;   // pre written here, converted in place

    const size_t XB = (size_t)N_TOK * D_IN * 2;    // bf16 x - b_dec
    const size_t WB = (size_t)D_SAE * D_IN * 2;    // bf16 W_enc
    const size_t WT = (size_t)D_SAE * D_IN * 4;    // f32 W_dec^T
    const size_t TI = (size_t)N_TOK * TOPK * 4;
    char* ws = (char*)d_ws;
    unsigned short* xb   = (unsigned short*)ws;
    unsigned short* wb   = (unsigned short*)(ws + XB);
    float*          Wt   = (float*)(ws + XB + WB);
    int*            tIdx = (int*)(ws + XB + WB + WT);
    float*          tVal = (float*)(ws + XB + WB + WT + TI);

    k_prep_x<<<(N_TOK * D_IN / 4 + 255) / 256, 256, 0, stream>>>(x, b_dec, xb);
    k_prep_w<<<((size_t)D_SAE * D_IN / 4 + 255) / 256, 256, 0, stream>>>(W_enc, wb);
    k_transpose<<<dim3(D_SAE / 32, D_IN / 32), dim3(32, 8), 0, stream>>>(W_dec, Wt);
    k_encode_gemm<<<dim3(D_SAE / BN, N_TOK / BM), 256, 0, stream>>>(xb, wb, b_enc, acts);
    k_topk<<<N_TOK, 256, 0, stream>>>(acts, tIdx, tVal);
    k_decode<<<N_TOK, 256, 0, stream>>>(Wt, b_dec, tIdx, tVal, recon);
}
